// multihead_attention_21835613733419
// MI455X (gfx1250) — matrix operand order repaired and checked by reading, not compiled, not run on hardware
//
#include <hip/hip_runtime.h>
#include <hip/hip_bf16.h>
#include <math.h>

// ---------------------------------------------------------------------------
// MI455X (gfx1250) bf16 WMMA implementation.
//   cvt_kernel : fc_w f32 -> bf16
//   attn_kernel: per-token head-mixing attention (wave32/token), 30 WMMAs/token
//   fc_kernel  : 16384x1024x1024 GEMM; W tile async-staged to LDS and shared
//                by 4 waves x 2 row-tiles each (B-fragment reuse).
// ---------------------------------------------------------------------------

typedef __attribute__((ext_vector_type(16))) __bf16        v16bf;
typedef __attribute__((ext_vector_type(8)))  float         v8f;
typedef __attribute__((ext_vector_type(4)))  unsigned int  v4u;

union Frag {
    v16bf v;
    v4u   q[2];   // two 16-byte chunks (8 bf16 each)
};

__device__ __forceinline__ unsigned short f2bf(float f) {
    unsigned int u = __float_as_uint(f);
    unsigned int r = u + 0x7FFFu + ((u >> 16) & 1u);   // round-to-nearest-even
    return (unsigned short)(r >> 16);
}

#define HEADS 16
#define HDIM  64
#define EMBED 1024
#define SEQ   4096
#define NB    4

// Convert f32 -> bf16 (for fc_w)
__global__ __launch_bounds__(256) void cvt_kernel(const float* __restrict__ src,
                                                  unsigned short* __restrict__ dst,
                                                  int n) {
    int i = blockIdx.x * 256 + threadIdx.x;
    if (i < n) dst[i] = f2bf(src[i]);
}

// Stage a token's 16x64 f32 tile into LDS as bf16 (row-major, 64 halves/row)
__device__ __forceinline__ void stage_token(const float* __restrict__ src,
                                            unsigned short* dst, int lane) {
    const float4* p = (const float4*)src;
#pragma unroll
    for (int i = 0; i < 8; ++i) {
        float4 x = p[lane + 32 * i];
        unsigned int p0 = (unsigned int)f2bf(x.x) | ((unsigned int)f2bf(x.y) << 16);
        unsigned int p1 = (unsigned int)f2bf(x.z) | ((unsigned int)f2bf(x.w) << 16);
        uint2 pk; pk.x = p0; pk.y = p1;
        *(uint2*)&dst[(lane + 32 * i) * 4] = pk;
    }
}

// C(16x16 tile at cols [d0,d0+16)) = X(16x64) @ W(64x64)^T  via two K=32 WMMAs.
__device__ __forceinline__ v8f proj_tile(const unsigned short* X,
                                         const unsigned short* W,
                                         int d0, int lm, int half) {
    v8f acc = {};
#pragma unroll
    for (int kb = 0; kb < 2; ++kb) {
        Frag a, b;
        const int abase = lm * 64 + kb * 32 + 8 * half;
        a.q[0] = *(const v4u*)&X[abase];
        a.q[1] = *(const v4u*)&X[abase + 16];
        const int bbase = (d0 + lm) * 64 + kb * 32 + 8 * half;
        b.q[0] = *(const v4u*)&W[bbase];
        b.q[1] = *(const v4u*)&W[bbase + 16];
        acc = __builtin_amdgcn_wmma_f32_16x16x32_bf16(false, a.v, false, b.v,
                                                      (short)0, acc, false, false);
    }
    return acc;
}

__global__ __launch_bounds__(128) void attn_kernel(
    const float* __restrict__ q, const float* __restrict__ k,
    const float* __restrict__ v, const int* __restrict__ mask,
    const float* __restrict__ qw, const float* __restrict__ kw,
    const float* __restrict__ vw, unsigned short* __restrict__ inter) {

    __shared__ __attribute__((aligned(16))) unsigned short Wq[64 * 64];
    __shared__ __attribute__((aligned(16))) unsigned short Wk[64 * 64];
    __shared__ __attribute__((aligned(16))) unsigned short Wv[64 * 64];
    __shared__ __attribute__((aligned(16))) unsigned short stage[4][16 * 64 + 8];
    __shared__ __attribute__((aligned(16))) unsigned short q4s[4][16 * 64 + 8];
    __shared__ __attribute__((aligned(16))) unsigned short k4s[4][16 * 64 + 8];
    __shared__ __attribute__((aligned(16))) unsigned short vTs[4][64 * 16 + 8];
    __shared__ __attribute__((aligned(16))) unsigned short wts[4][16 * 16 + 8];

    const int tid  = threadIdx.x;
    const int wid  = tid >> 5;
    const int lane = tid & 31;
    const int lm   = lane & 15;
    const int half = lane >> 4;

    // convert projection weights once per block (shared across 4 waves)
    for (int i = tid; i < 64 * 64; i += 128) {
        Wq[i] = f2bf(qw[i]);
        Wk[i] = f2bf(kw[i]);
        Wv[i] = f2bf(vw[i]);
    }
    __syncthreads();

    const int t  = blockIdx.x * 4 + wid;        // token id, grid covers exactly N*L
    const int nb = t >> 12;                     // / SEQ
    const int l  = t & (SEQ - 1);
    const size_t tokoff = ((size_t)nb * SEQ + l) * EMBED;

    // ---- Q projection ----
    stage_token(q + tokoff, stage[wid], lane);
    __syncthreads();
#pragma unroll
    for (int ti = 0; ti < 4; ++ti) {
        v8f acc = proj_tile(stage[wid], Wq, ti * 16, lm, half);
#pragma unroll
        for (int j = 0; j < 8; ++j)
            q4s[wid][(j + 8 * half) * 64 + ti * 16 + lm] = f2bf(acc[j]);
    }
    __syncthreads();

    // ---- K projection ----
    stage_token(k + tokoff, stage[wid], lane);
    __syncthreads();
#pragma unroll
    for (int ti = 0; ti < 4; ++ti) {
        v8f acc = proj_tile(stage[wid], Wk, ti * 16, lm, half);
#pragma unroll
        for (int j = 0; j < 8; ++j)
            k4s[wid][(j + 8 * half) * 64 + ti * 16 + lm] = f2bf(acc[j]);
    }
    __syncthreads();

    // ---- V projection (stored transposed: vT[d][g] = v4[g][d]) ----
    stage_token(v + tokoff, stage[wid], lane);
    __syncthreads();
#pragma unroll
    for (int ti = 0; ti < 4; ++ti) {
        v8f acc = proj_tile(stage[wid], Wv, ti * 16, lm, half);
#pragma unroll
        for (int j = 0; j < 8; ++j)
            vTs[wid][(ti * 16 + lm) * 16 + (j + 8 * half)] = f2bf(acc[j]);
    }
    __syncthreads();

    // ---- logits = q4 @ k4^T * 1/sqrt(64)  (16x16, K=64) ----
    v8f lg = {};
#pragma unroll
    for (int kb = 0; kb < 2; ++kb) {
        Frag a, b;
        const int abase = lm * 64 + kb * 32 + 8 * half;
        a.q[0] = *(const v4u*)&q4s[wid][abase];
        a.q[1] = *(const v4u*)&q4s[wid][abase + 16];
        const int bbase = lm * 64 + kb * 32 + 8 * half;   // B[k][g] = k4[g][k]
        b.q[0] = *(const v4u*)&k4s[wid][bbase];
        b.q[1] = *(const v4u*)&k4s[wid][bbase + 16];
        lg = __builtin_amdgcn_wmma_f32_16x16x32_bf16(false, a.v, false, b.v,
                                                     (short)0, lg, false, false);
    }

    // ---- masked softmax over g (16 lanes per half hold one row) ----
    const int* mtok = mask + ((size_t)nb * SEQ + l) * 256;
    float wv[8];
#pragma unroll
    for (int j = 0; j < 8; ++j) {
        const int m = j + 8 * half;
        float x = lg[j] * 0.125f;
        if (mtok[m * 16 + lm] == 0) x = -INFINITY;
        float mx = x;
#pragma unroll
        for (int off = 1; off < 16; off <<= 1)
            mx = fmaxf(mx, __shfl_xor(mx, off, 32));
        float e = __expf(x - mx);
        float s = e;
#pragma unroll
        for (int off = 1; off < 16; off <<= 1)
            s += __shfl_xor(s, off, 32);
        wv[j] = e / s;
    }
#pragma unroll
    for (int j = 0; j < 8; ++j)
        wts[wid][(j + 8 * half) * 16 + lm] = f2bf(wv[j]);
    __syncthreads();

    // ---- out4 = weights(16x16) @ v4(16x64), K padded 16->32 with zeros ----
    const int  l2base  = l >> 4;
    const int  colbase = (l & 15) * 64;
    const v4u  z = {0u, 0u, 0u, 0u};
#pragma unroll
    for (int ti = 0; ti < 4; ++ti) {
        Frag a, b;
        a.q[0] = *(const v4u*)&wts[wid][lm * 16 + 8 * half];
        a.q[1] = z;                                   // k >= 16 is zero
        v4u b0 = *(const v4u*)&vTs[wid][(ti * 16 + lm) * 16];
        v4u b1 = *(const v4u*)&vTs[wid][(ti * 16 + lm) * 16 + 8];
        b.q[0] = half ? b1 : b0;                      // lanes 16-31 hold k 8..15
        b.q[1] = z;                                   // k >= 16 is zero
        v8f o = {};
        o = __builtin_amdgcn_wmma_f32_16x16x32_bf16(false, a.v, false, b.v,
                                                    (short)0, o, false, false);
        // scrambled reshape: (n, h, l, d) -> row l2 = h*256 + l/16, col = (l%16)*64 + d
#pragma unroll
        for (int j = 0; j < 8; ++j) {
            const int m   = j + 8 * half;
            const int row = m * 256 + l2base;
            inter[((size_t)nb * SEQ + row) * EMBED + colbase + ti * 16 + lm] = f2bf(o[j]);
        }
    }
}

// out(16384x1024) = inter(bf16) @ fc_w^T + fc_b.
// Block: 4 waves share one 64-col W tile (LDS, async-staged in 32KB chunks);
// each wave owns two 16-row tiles -> B fragments reused across 8 WMMAs/kstep.
__global__ __launch_bounds__(128) void fc_kernel(
    const unsigned short* __restrict__ A, const unsigned short* __restrict__ W,
    const float* __restrict__ bias, float* __restrict__ out) {

    __shared__ __attribute__((aligned(16))) unsigned short Wl[64 * 256];  // 32 KB chunk

    const int tid  = threadIdx.x;
    const int wid  = tid >> 5;
    const int lane = tid & 31;
    const int lm   = lane & 15;
    const int half = lane >> 4;

    const int colblk = blockIdx.x & 15;            // 16 column blocks of 64
    const int rowgrp = blockIdx.x >> 4;            // 128 row groups of 8 tiles
    const int n0     = colblk * 64;
    const int r0     = (rowgrp * 8 + wid * 2) * 16;   // this wave: rows [r0,r0+32)

    v8f acc0[4] = {};
    v8f acc1[4] = {};
    const size_t arow0 = (size_t)(r0 + lm) * EMBED;
    const size_t arow1 = arow0 + (size_t)16 * EMBED;
    const unsigned long long wbase = (unsigned long long)(size_t)W;

    for (int c = 0; c < 4; ++c) {
        // async-stage W[n0..n0+64) x k[c*256 .. c*256+256) into LDS (2048 x 16B)
#pragma unroll
        for (int i = 0; i < 16; ++i) {
            const int idx = tid + 128 * i;
            const int row = idx >> 5;              // 0..63
            const int seg = idx & 31;              // 32 x 16B per row
            const unsigned ldsa = (unsigned)(size_t)&Wl[row * 256 + seg * 8];
            const unsigned voff = (unsigned)(((n0 + row) * EMBED + c * 256 + seg * 8) * 2);
            asm volatile("global_load_async_to_lds_b128 %0, %1, %2 offset:0"
                         :: "v"(ldsa), "v"(voff), "s"(wbase) : "memory");
        }
        asm volatile("s_wait_asynccnt 0x0" ::: "memory");
        __syncthreads();

#pragma unroll
        for (int kk = 0; kk < 256; kk += 32) {
            const int ks = c * 256 + kk;
            Frag a0, a1;
            a0.q[0] = *(const v4u*)&A[arow0 + ks + 8 * half];
            a0.q[1] = *(const v4u*)&A[arow0 + ks + 16 + 8 * half];
            a1.q[0] = *(const v4u*)&A[arow1 + ks + 8 * half];
            a1.q[1] = *(const v4u*)&A[arow1 + ks + 16 + 8 * half];
#pragma unroll
            for (int ti = 0; ti < 4; ++ti) {
                Frag b;
                const int brow = (ti * 16 + lm) * 256 + kk + 8 * half;
                b.q[0] = *(const v4u*)&Wl[brow];
                b.q[1] = *(const v4u*)&Wl[brow + 16];
                acc0[ti] = __builtin_amdgcn_wmma_f32_16x16x32_bf16(
                    false, a0.v, false, b.v, (short)0, acc0[ti], false, false);
                acc1[ti] = __builtin_amdgcn_wmma_f32_16x16x32_bf16(
                    false, a1.v, false, b.v, (short)0, acc1[ti], false, false);
            }
        }
        __syncthreads();   // protect Wl before next chunk overwrites it
    }

#pragma unroll
    for (int ti = 0; ti < 4; ++ti) {
        const int n  = n0 + ti * 16 + lm;
        const float bv = bias[n];
#pragma unroll
        for (int j = 0; j < 8; ++j) {
            out[(size_t)(r0 + j + 8 * half) * EMBED + n]      = acc0[ti][j] + bv;
            out[(size_t)(r0 + 16 + j + 8 * half) * EMBED + n] = acc1[ti][j] + bv;
        }
    }
}

extern "C" void kernel_launch(void* const* d_in, const int* in_sizes, int n_in,
                              void* d_out, int out_size, void* d_ws, size_t ws_size,
                              hipStream_t stream) {
    const float* q    = (const float*)d_in[0];
    const float* k    = (const float*)d_in[1];
    const float* v    = (const float*)d_in[2];
    const int*   mask = (const int*)d_in[3];
    const float* qw   = (const float*)d_in[4];
    const float* kw   = (const float*)d_in[5];
    const float* vw   = (const float*)d_in[6];
    const float* fcw  = (const float*)d_in[7];
    const float* fcb  = (const float*)d_in[8];
    float* out = (float*)d_out;

    // workspace layout: [0, 2MB) fc_w bf16 ; [2MB, 34MB) scrambled intermediate bf16
    unsigned short* fcw_bf = (unsigned short*)d_ws;
    unsigned short* inter  = fcw_bf + (size_t)EMBED * EMBED;

    cvt_kernel<<<(EMBED * EMBED) / 256, 256, 0, stream>>>(fcw, fcw_bf, EMBED * EMBED);
    attn_kernel<<<(NB * SEQ) / 4, 128, 0, stream>>>(q, k, v, mask, qw, kw, vw, inter);
    // 128 row groups x 16 col blocks
    fc_kernel<<<128 * 16, 128, 0, stream>>>(inter, fcw_bf, fcb, out);
}
